// MPConvSentenceEncoderLayer_58849641890377
// MI455X (gfx1250) — matrix operand order repaired and checked by reading, not compiled, not run on hardware
//
#include <hip/hip_runtime.h>
#include <math.h>

// MPConvSentenceEncoderLayer for MI455X (gfx1250, wave32, WMMA bf16).
//
// out = LN( x + gelu( mp2( gelu( mp1(x) ) ) ) ) with mp = x + conv{1,3,5}(x) + b
// Key transform: x + conv1+conv3+conv5 == single k=5 conv with tap-fused weights
// (9*C^2 -> 5*C^2 MACs). Each conv-layer becomes a GEMM with K = 5*768, mapped
// onto v_wmma_f32_16x16x32_bf16 (bf16 A/B, f32 accumulate).
//
// Workspace (d_ws) partition (~204 MB):
//   packed weights (bf16, WMMA-A swizzled) | xT | h1T (bf16 [B][T+4][C], 2-row
//   zero halo => no bounds checks in K loop) | h1/h2 natural bf16 [B][C][T].

#define B_  32
#define C_  768
#define T_  1024
#define T4_ (T_ + 4)
#define KT_ 24   // 768 / 32  K-chunks per tap
#define MT_ 48   // 768 / 16  M-tiles

typedef __attribute__((ext_vector_type(16))) __bf16 v16bf;
typedef __attribute__((ext_vector_type(8)))  float  v8f;

__device__ __forceinline__ float gelu_exact(float v) {
    return 0.5f * v * (1.0f + erff(v * 0.70710678118654752f));
}

// ---------------------------------------------------------------------------
// Fuse {w1,w3,w5} into 5 taps and pre-swizzle into this kernel's contiguous-k A-operand lane
// layout: tile = 16(M=co) x 32(K=ci) bf16; lane l (l&15 = M row) holds
// K = base + h, base = (l>=16 ? 16 : 0). Each lane's 16
// halves are stored contiguously -> a single 32B global load per lane.
// packed index: ((((layer*5+tap)*KT+kt)*MT+mt)*32 + lane)*16 + h
// ---------------------------------------------------------------------------
__global__ void pack_weights(const float* __restrict__ w1_1, const float* __restrict__ w1_3,
                             const float* __restrict__ w1_5, const float* __restrict__ w2_1,
                             const float* __restrict__ w2_3, const float* __restrict__ w2_5,
                             __bf16* __restrict__ packed) {
    int idx = blockIdx.x * 256 + threadIdx.x;   // total = 2*5*24*48*512
    int h  = idx & 15;
    int l  = (idx >> 4) & 31;
    int r  = idx >> 9;
    int mt = r % MT_;  r /= MT_;
    int kt = r % KT_;  r /= KT_;
    int tap = r % 5;
    int layer = r / 5;

    int klocal = ((l >> 4) << 4) + h;
    int co = mt * 16 + (l & 15);
    int ci = kt * 32 + klocal;

    const float* wk1 = layer ? w2_1 : w1_1;
    const float* wk3 = layer ? w2_3 : w1_3;
    const float* wk5 = layer ? w2_5 : w1_5;

    int cc = co * C_ + ci;
    float v = wk5[cc * 5 + tap];                  // k=5 contributes to all taps
    if (tap >= 1 && tap <= 3) v += wk3[cc * 3 + (tap - 1)];
    if (tap == 2)             v += wk1[cc];
    packed[idx] = (__bf16)v;
}

// Zero the 2-row halos of both transposed activation buffers.
__global__ void zero_halo(__bf16* __restrict__ xT, __bf16* __restrict__ h1T) {
    int idx = blockIdx.x * 256 + threadIdx.x;    // total = B*4*C
    if (idx >= B_ * 4 * C_) return;
    int c = idx % C_;
    int r = (idx / C_) & 3;
    int b = idx / (4 * C_);
    int row = (r < 2) ? r : (T_ + r);            // rows 0,1,T+2,T+3
    size_t off = ((size_t)(b * T4_ + row)) * C_ + c;
    xT[off]  = (__bf16)0.0f;
    h1T[off] = (__bf16)0.0f;
}

// x [B][C][T] fp32 -> xT [B][T+4][C] bf16 (interior rows), LDS-tiled transpose.
__global__ void transpose_cast(const float* __restrict__ x, __bf16* __restrict__ xT) {
    __shared__ float tile[32][33];
    int t0 = blockIdx.x * 32, c0 = blockIdx.y * 32, b = blockIdx.z;
    int tid = threadIdx.x;
#pragma unroll
    for (int it = 0; it < 4; ++it) {
        int idx = it * 256 + tid;
        int cc = idx >> 5, tt = idx & 31;        // coalesced along t
        tile[cc][tt] = x[((size_t)(b * C_ + c0 + cc)) * T_ + t0 + tt];
    }
    __syncthreads();
#pragma unroll
    for (int it = 0; it < 4; ++it) {
        int idx = it * 256 + tid;
        int tt = idx >> 5, cc = idx & 31;        // contiguous along c
        xT[((size_t)(b * T4_ + t0 + tt + 2)) * C_ + c0 + cc] = (__bf16)tile[cc][tt];
    }
}

// ---------------------------------------------------------------------------
// Conv-as-GEMM with WMMA. Block = 256 threads = 8 waves (2 M-groups x 4
// N-groups); each wave owns 2x2 16x16 f32 accumulators -> block tile 64co x
// 128t. K loop: 24 ci-chunks x 5 taps; B-tiles read straight from the
// transposed bf16 activations (halo rows make every read in-bounds).
// Epilogue fuses residual + bias + exact GELU, writing a natural-layout bf16
// copy (coalesced) and optionally the next layer's transposed input.
// ---------------------------------------------------------------------------
template <bool RES_F32, bool WRITE_TR>
__global__ __launch_bounds__(256)
void mpconv_gemm(const __bf16* __restrict__ inT,    // [B][T4][C] bf16
                 const void*   __restrict__ resid,  // [B][C][T] fp32 or bf16
                 const __bf16* __restrict__ wpack,  // [5][KT][MT][512]
                 const float*  __restrict__ bias,   // [C]
                 __bf16* __restrict__ outNat,       // [B][C][T]
                 __bf16* __restrict__ outT)         // [B][T4][C] (or null)
{
    const int tid  = threadIdx.x;
    const int lane = tid & 31;
    const int wave = tid >> 5;
    const int wm = wave & 1, wn = wave >> 1;
    const int b   = blockIdx.z;
    const int co0 = blockIdx.y * 64 + wm * 32;   // wave: co0..co0+31 (2 M-tiles)
    const int t0  = blockIdx.x * 128 + wn * 32;  // wave: t0..t0+31  (2 N-tiles)
    const int Nl  = lane & 15;
    const int hi  = lane >> 4;

    v8f acc[2][2] = {};

    const v16bf* wp = (const v16bf*)wpack;       // 32 vectors per 16x32 tile
    const v16bf* xp = (const v16bf*)inT;         // 48 vectors per C row
    const int mtg0 = co0 >> 4;
    const size_t xrowv = C_ / 16;                // 48
    const size_t xbase = (size_t)b * T4_ * xrowv;

    for (int kt = 0; kt < KT_; ++kt) {
#pragma unroll
        for (int tap = 0; tap < 5; ++tap) {
            size_t abase = ((size_t)(tap * KT_ + kt)) * MT_ * 32 + lane;
            v16bf a0 = wp[abase + (size_t)(mtg0 + 0) * 32];
            v16bf a1 = wp[abase + (size_t)(mtg0 + 1) * 32];
            // B-tile: lane holds N=Nl, K = hi*16 .. hi*16+15 (contiguous ci).
            size_t rb = xbase + (size_t)(t0 + Nl + tap) * xrowv + kt * 2 + hi;
            v16bf b0 = xp[rb];
            v16bf b1 = xp[rb + 16 * xrowv];
            acc[0][0] = __builtin_amdgcn_wmma_f32_16x16x32_bf16(false, a0, false, b0, (short)0, acc[0][0], false, false);
            acc[0][1] = __builtin_amdgcn_wmma_f32_16x16x32_bf16(false, a0, false, b1, (short)0, acc[0][1], false, false);
            acc[1][0] = __builtin_amdgcn_wmma_f32_16x16x32_bf16(false, a1, false, b0, (short)0, acc[1][0], false, false);
            acc[1][1] = __builtin_amdgcn_wmma_f32_16x16x32_bf16(false, a1, false, b1, (short)0, acc[1][1], false, false);
        }
    }

    // Epilogue: D element (lane, vgpr r) -> M = r + hi*8, N = Nl.
#pragma unroll
    for (int mi = 0; mi < 2; ++mi)
#pragma unroll
        for (int ni = 0; ni < 2; ++ni)
#pragma unroll
            for (int r = 0; r < 8; ++r) {
                int co = co0 + mi * 16 + r + hi * 8;
                int t  = t0  + ni * 16 + Nl;
                size_t off = ((size_t)(b * C_ + co)) * T_ + t;
                float res = RES_F32 ? ((const float*)resid)[off]
                                    : (float)((const __bf16*)resid)[off];
                float v = acc[mi][ni][r] + res + bias[co];
                float g = gelu_exact(v);
                outNat[off] = (__bf16)g;
                if (WRITE_TR)
                    outT[((size_t)(b * T4_ + t + 2)) * C_ + co] = (__bf16)g;
            }
}

// out = LayerNorm_C(x + h2) * gamma + beta.  Lane-per-t => coalesced along T.
__global__ __launch_bounds__(128)
void residual_ln(const float* __restrict__ x, const __bf16* __restrict__ h2,
                 const float* __restrict__ gamma, const float* __restrict__ beta,
                 float* __restrict__ out) {
    int b = blockIdx.y;
    int t = blockIdx.x * 128 + threadIdx.x;
    float s = 0.f, ss = 0.f;
    for (int c = 0; c < C_; ++c) {
        size_t off = ((size_t)(b * C_ + c)) * T_ + t;
        float v = x[off] + (float)h2[off];
        s += v; ss += v * v;
    }
    float mu  = s * (1.0f / C_);
    float var = ss * (1.0f / C_) - mu * mu;
    float inv = rsqrtf(var + 1e-5f);
    for (int c = 0; c < C_; ++c) {
        size_t off = ((size_t)(b * C_ + c)) * T_ + t;
        float v = x[off] + (float)h2[off];
        out[off] = (v - mu) * inv * gamma[c] + beta[c];
    }
}

extern "C" void kernel_launch(void* const* d_in, const int* in_sizes, int n_in,
                              void* d_out, int out_size, void* d_ws, size_t ws_size,
                              hipStream_t stream) {
    // setup_inputs() dict order: x, w1_1, w2_1, w1_3, w2_3, w1_5, w2_5,
    //                            b1, b2, ln_gamma, ln_beta
    const float* x    = (const float*)d_in[0];
    const float* w1_1 = (const float*)d_in[1];
    const float* w2_1 = (const float*)d_in[2];
    const float* w1_3 = (const float*)d_in[3];
    const float* w2_3 = (const float*)d_in[4];
    const float* w1_5 = (const float*)d_in[5];
    const float* w2_5 = (const float*)d_in[6];
    const float* b1   = (const float*)d_in[7];
    const float* b2   = (const float*)d_in[8];
    const float* lng  = (const float*)d_in[9];
    const float* lnb  = (const float*)d_in[10];

    // Workspace partition (~204 MB total).
    char* ws = (char*)d_ws;
    const size_t packedElems = 2ull * 5 * KT_ * MT_ * 512;      // both layers
    __bf16* packed = (__bf16*)ws;  ws += packedElems * 2;
    const size_t xTSz = (size_t)B_ * T4_ * C_ * 2;
    __bf16* xT  = (__bf16*)ws;     ws += xTSz;
    __bf16* h1T = (__bf16*)ws;     ws += xTSz;
    const size_t natSz = (size_t)B_ * C_ * T_ * 2;
    __bf16* h1n = (__bf16*)ws;     ws += natSz;
    __bf16* h2n = (__bf16*)ws;     ws += natSz;

    hipLaunchKernelGGL(pack_weights, dim3((unsigned)(packedElems / 256)), dim3(256), 0, stream,
                       w1_1, w1_3, w1_5, w2_1, w2_3, w2_5, packed);
    hipLaunchKernelGGL(zero_halo, dim3((B_ * 4 * C_ + 255) / 256), dim3(256), 0, stream, xT, h1T);
    hipLaunchKernelGGL(transpose_cast, dim3(T_ / 32, C_ / 32, B_), dim3(256), 0, stream, x, xT);

    // Layer 1: h1 = gelu(x + conv5(x, We1) + b1)   (residual read as fp32)
    hipLaunchKernelGGL((mpconv_gemm<true, true>), dim3(T_ / 128, C_ / 64, B_), dim3(256), 0, stream,
                       xT, (const void*)x, packed, b1, h1n, h1T);
    // Layer 2: h2 = gelu(h1 + conv5(h1, We2) + b2) (residual read as bf16)
    hipLaunchKernelGGL((mpconv_gemm<false, false>), dim3(T_ / 128, C_ / 64, B_), dim3(256), 0, stream,
                       h1T, (const void*)h1n, packed + 5ull * KT_ * MT_ * 512, b2, h2n, (__bf16*)nullptr);

    hipLaunchKernelGGL(residual_ln, dim3(T_ / 128, B_), dim3(128), 0, stream,
                       x, h2n, lng, lnb, (float*)d_out);
}
